// PostNormAttentionBlock_52905407152493
// MI455X (gfx1250) — matrix operand order repaired and checked by reading, not compiled, not run on hardware
//
// PostNorm transformer block for MI455X (gfx1250), wave32 + WMMA bf16 (hi/lo split x3).
// bf16 hi/lo planes -> ds_load_b128 straight into WMMA fragments (no unpack VALU).
// Tile staging uses CDNA5 async global->LDS (ASYNCcnt) with double buffering.
#include <hip/hip_runtime.h>
#include <cstdint>
#include <cmath>

#define DEV __device__ __forceinline__
typedef unsigned short u16;
typedef int v4i __attribute__((vector_size(16)));

typedef __attribute__((ext_vector_type(16))) __bf16 v16bf;
typedef __attribute__((ext_vector_type(8)))  float  v8f;

union Frag16 { v16bf v; unsigned u[8]; };

#if __has_builtin(__builtin_amdgcn_global_load_async_to_lds_b128)
#define HAS_ASYNC_LDS 1
#else
#define HAS_ASYNC_LDS 0
#endif

DEV void async_copy16(const u16* g, u16* l) {
#if HAS_ASYNC_LDS
  __builtin_amdgcn_global_load_async_to_lds_b128(
      (__attribute__((address_space(1))) v4i*)(void*)g,
      (__attribute__((address_space(3))) v4i*)(void*)l, 0, 0);
#else
  *(uint4*)l = *(const uint4*)g;
#endif
}
DEV void wait_async() {
#if HAS_ASYNC_LDS
#if __has_builtin(__builtin_amdgcn_s_wait_asynccnt)
  __builtin_amdgcn_s_wait_asynccnt(0);
#else
  asm volatile("s_wait_asynccnt 0x0" ::: "memory");
#endif
#endif
}

// ---------- bf16 hi/lo split ----------
DEV unsigned bf16_rne(float x) {
  unsigned u = __float_as_uint(x);
  return (u + 0x7FFFu + ((u >> 16) & 1u)) >> 16;
}
DEV void split2(float x, u16& h, u16& l) {
  unsigned hi = bf16_rne(x);
  float r = x - __uint_as_float(hi << 16);
  h = (u16)hi;
  l = (u16)bf16_rne(r);
}
DEV unsigned pack_hilo(float x) {  // interleaved form (attention P tile only)
  u16 h, l;
  split2(x, h, l);
  return (unsigned)h | ((unsigned)l << 16);
}
DEV void build_frags(const unsigned p[16], Frag16& hi, Frag16& lo) {
#pragma unroll
  for (int i = 0; i < 8; ++i) {
    unsigned a = p[2 * i], b = p[2 * i + 1];
    hi.u[i] = (a & 0xFFFFu) | (b << 16);
    lo.u[i] = (a >> 16) | (b & 0xFFFF0000u);
  }
}

// ---------- fragment loaders from LDS bf16 planes (no unpack VALU) ----------
DEV void load_afrag16(const u16* lds, int m0, int ldk, int koff, Frag16& f) {
  int lane = threadIdx.x & 31;
  const u16* row = lds + (m0 + (lane & 15)) * ldk + koff + ((lane >> 4) & 1) * 8;
  *(uint4*)&f.u[0] = *(const uint4*)row;
  *(uint4*)&f.u[4] = *(const uint4*)(row + 16);
}
DEV void load_bfrag16(const u16* lds, int n0, int ldk, int koff, Frag16& f) {
  int lane = threadIdx.x & 31;
  const u16* row = lds + (n0 + (lane & 15)) * ldk + koff + ((lane >> 4) & 1) * 8;
  *(uint4*)&f.u[0] = *(const uint4*)row;
  *(uint4*)&f.u[4] = *(const uint4*)(row + 16);
}
DEV void load_afrag32(const unsigned* lds, int m0, int ldk, int koff, Frag16& hi, Frag16& lo) {
  int lane = threadIdx.x & 31;
  const unsigned* row = lds + (m0 + (lane & 15)) * ldk + koff + ((lane >> 4) & 1) * 8;
  unsigned p[16];
#pragma unroll
  for (int j = 0; j < 8; ++j) p[j] = row[j];
#pragma unroll
  for (int j = 0; j < 8; ++j) p[8 + j] = row[16 + j];
  build_frags(p, hi, lo);
}

// 3-term split product: (Ah+Al)(Bh+Bl) ~= AhBh + AhBl + AlBh
DEV v8f wmma3(const Frag16& ah, const Frag16& al, const Frag16& bh, const Frag16& bl, v8f c) {
  c = __builtin_amdgcn_wmma_f32_16x16x32_bf16(false, ah.v, false, bh.v, (short)0, c, false, false);
  c = __builtin_amdgcn_wmma_f32_16x16x32_bf16(false, ah.v, false, bl.v, (short)0, c, false, false);
  c = __builtin_amdgcn_wmma_f32_16x16x32_bf16(false, al.v, false, bh.v, (short)0, c, false, false);
  return c;
}

// ---------- elementwise pack kernels ----------
__global__ void pack_kernel(const float* __restrict__ in, u16* __restrict__ hi,
                            u16* __restrict__ lo, int n) {
  int i = blockIdx.x * 256 + threadIdx.x;
  if (i < n) { u16 h, l; split2(in[i], h, l); hi[i] = h; lo[i] = l; }
}
__global__ void packT_kernel(const float* __restrict__ in, u16* __restrict__ hi,
                             u16* __restrict__ lo, int K, int N) {
  int i = blockIdx.x * 256 + threadIdx.x;
  if (i < K * N) {
    int n = i % N, k = i / N;
    u16 h, l; split2(in[i], h, l);
    hi[(size_t)n * K + k] = h;
    lo[(size_t)n * K + k] = l;
  }
}

// ---------- generic WMMA GEMM: C = epi(A(MxK) @ Bt(NxK)^T + bias) ----------
enum { EPI_QKV = 0, EPI_VT = 1, EPI_GELU = 2, EPI_RES = 3 };

template <int EPI>
__global__ __launch_bounds__(256) void gemm_bf16x3(
    const u16* __restrict__ Ahi, const u16* __restrict__ Alo,
    const u16* __restrict__ Bhi, const u16* __restrict__ Blo,
    const float* __restrict__ bias, const float* __restrict__ res,
    void* __restrict__ outv, u16* __restrict__ out_lo,
    int M, int N, int K, float scale) {
  __shared__ u16 lAh[2][128 * 40], lAl[2][128 * 40];
  __shared__ u16 lBh[2][128 * 40], lBl[2][128 * 40];
  const int tid = threadIdx.x, lane = tid & 31, wave = tid >> 5;
  const int wm = wave >> 2, wn = wave & 3;  // 2 x 4 wave grid, each 64M x 32N
  const int bm = blockIdx.x * 128, bn = blockIdx.y * 128;

  auto stage = [&](int k0, int buf) {
#if HAS_ASYNC_LDS
#pragma unroll
    for (int i = 0; i < 2; ++i) {
      int s = i * 256 + tid, r = s >> 1, c = (s & 1) << 4;
      async_copy16(Ahi + (size_t)(bm + r) * K + k0 + c, &lAh[buf][r * 40 + c]);
      async_copy16(Alo + (size_t)(bm + r) * K + k0 + c, &lAl[buf][r * 40 + c]);
      async_copy16(Bhi + (size_t)(bn + r) * K + k0 + c, &lBh[buf][r * 40 + c]);
      async_copy16(Blo + (size_t)(bn + r) * K + k0 + c, &lBl[buf][r * 40 + c]);
    }
#else
    uint4 t[8];
#pragma unroll
    for (int i = 0; i < 2; ++i) {
      int s = i * 256 + tid, r = s >> 1, c = (s & 1) << 4;
      t[4 * i + 0] = *(const uint4*)(Ahi + (size_t)(bm + r) * K + k0 + c);
      t[4 * i + 1] = *(const uint4*)(Alo + (size_t)(bm + r) * K + k0 + c);
      t[4 * i + 2] = *(const uint4*)(Bhi + (size_t)(bn + r) * K + k0 + c);
      t[4 * i + 3] = *(const uint4*)(Blo + (size_t)(bn + r) * K + k0 + c);
    }
#pragma unroll
    for (int i = 0; i < 2; ++i) {
      int s = i * 256 + tid, r = s >> 1, c = (s & 1) << 4;
      *(uint4*)&lAh[buf][r * 40 + c] = t[4 * i + 0];
      *(uint4*)&lAl[buf][r * 40 + c] = t[4 * i + 1];
      *(uint4*)&lBh[buf][r * 40 + c] = t[4 * i + 2];
      *(uint4*)&lBl[buf][r * 40 + c] = t[4 * i + 3];
    }
#endif
  };

  v8f acc[4][2];
  v8f zf = {};
#pragma unroll
  for (int mi = 0; mi < 4; ++mi)
#pragma unroll
    for (int ni = 0; ni < 2; ++ni) acc[mi][ni] = zf;

  stage(0, 0);
  int cur = 0;
  for (int k0 = 0; k0 < K; k0 += 32) {
    wait_async();
    __syncthreads();
    if (k0 + 32 < K) stage(k0 + 32, cur ^ 1);  // prefetch next tile into other buffer
    Frag16 ah[4], al[4], bh[2], bl[2];
#pragma unroll
    for (int mi = 0; mi < 4; ++mi) {
      load_afrag16(lAh[cur], wm * 64 + mi * 16, 40, 0, ah[mi]);
      load_afrag16(lAl[cur], wm * 64 + mi * 16, 40, 0, al[mi]);
    }
#pragma unroll
    for (int ni = 0; ni < 2; ++ni) {
      load_bfrag16(lBh[cur], wn * 32 + ni * 16, 40, 0, bh[ni]);
      load_bfrag16(lBl[cur], wn * 32 + ni * 16, 40, 0, bl[ni]);
    }
#pragma unroll
    for (int mi = 0; mi < 4; ++mi)
#pragma unroll
      for (int ni = 0; ni < 2; ++ni) acc[mi][ni] = wmma3(ah[mi], al[mi], bh[ni], bl[ni], acc[mi][ni]);
    cur ^= 1;
  }

  // epilogue. C layout: VGPR r -> row r + 8*(lane>>4), col lane&15
#pragma unroll
  for (int mi = 0; mi < 4; ++mi)
#pragma unroll
    for (int ni = 0; ni < 2; ++ni)
#pragma unroll
      for (int r = 0; r < 8; ++r) {
        int m = bm + wm * 64 + mi * 16 + r + 8 * (lane >> 4);
        int n = bn + wn * 32 + ni * 16 + (lane & 15);
        float v = acc[mi][ni][r] + bias[n];
        if constexpr (EPI == EPI_RES) {
          ((float*)outv)[(size_t)m * N + n] = v + res[(size_t)m * N + n];
        } else if constexpr (EPI == EPI_GELU) {
          float g = 0.5f * v * (1.0f + erff(v * 0.70710678118654752f));
          u16 h_, l_; split2(g, h_, l_);
          size_t idx = (size_t)m * N + n;
          ((u16*)outv)[idx] = h_; out_lo[idx] = l_;
        } else if constexpr (EPI == EPI_QKV) {
          v *= scale;  // folds 1/sqrt(D) for Q
          int b = m >> 11, s = m & 2047, h = n >> 6, d = n & 63;
          size_t idx = ((((size_t)b * 8 + h) << 11) + s) * 64 + d;
          u16 h_, l_; split2(v, h_, l_);
          ((u16*)outv)[idx] = h_; out_lo[idx] = l_;
        } else {  // EPI_VT: out[b][h][d][s]
          int b = m >> 11, s = m & 2047, h = n >> 6, d = n & 63;
          size_t idx = ((((size_t)b * 8 + h) * 64 + d) << 11) + s;
          u16 h_, l_; split2(v, h_, l_);
          ((u16*)outv)[idx] = h_; out_lo[idx] = l_;
        }
      }
}

// ---------- flash attention: one WG per (b,h) x 128 query rows ----------
__global__ __launch_bounds__(256) void attn_kernel(
    const u16* __restrict__ qhi, const u16* __restrict__ qlo,
    const u16* __restrict__ khi, const u16* __restrict__ klo,
    const u16* __restrict__ vthi, const u16* __restrict__ vtlo,
    u16* __restrict__ yhi, u16* __restrict__ ylo) {
  // Double-buffered K planes (2x 2x 128x72 u16) + V^T planes (2x 2x 64x136 u16)
  // + P interleaved u32 (128x132); Q staging planes overlay the P region.
  // Pointers computed at use sites (no static-initialized pointer arrays).
  __shared__ __align__(16) unsigned char smem[210944];
  auto KH = [&](int buf) -> u16* { return (u16*)(smem + 36864 * buf); };
  auto KL = [&](int buf) -> u16* { return (u16*)(smem + 36864 * buf) + 128 * 72; };
  auto VH = [&](int buf) -> u16* { return (u16*)(smem + 73728 + 34816 * buf); };
  auto VL = [&](int buf) -> u16* { return (u16*)(smem + 73728 + 34816 * buf) + 64 * 136; };
  unsigned* lP = (unsigned*)(smem + 143360);
  u16* lQh = (u16*)(smem + 143360);
  u16* lQl = lQh + 128 * 72;

  const int tid = threadIdx.x, lane = tid & 31, wave = tid >> 5;
  const int bh = blockIdx.y, b = bh >> 3, h = bh & 7;
  const int qs0 = blockIdx.x * 128;
  const size_t base = (size_t)bh * (2048 * 64);
  const u16* qh_ = qhi + base;  const u16* ql_ = qlo + base;
  const u16* kh_ = khi + base;  const u16* kl_ = klo + base;
  const u16* vh_ = vthi + base; const u16* vl_ = vtlo + base;

  auto stageKV = [&](int t0, int buf) {
    u16* dKh = KH(buf); u16* dKl = KL(buf);
    u16* dVh = VH(buf); u16* dVl = VL(buf);
#pragma unroll
    for (int i = 0; i < 4; ++i) {
      int s = i * 256 + tid;
      { int r = s >> 3, c = (s & 7) << 3;
        async_copy16(kh_ + (size_t)(t0 + r) * 64 + c, dKh + r * 72 + c);
        async_copy16(kl_ + (size_t)(t0 + r) * 64 + c, dKl + r * 72 + c); }
      { int d = s >> 4, c = (s & 15) << 3;
        async_copy16(vh_ + (size_t)d * 2048 + t0 + c, dVh + d * 136 + c);
        async_copy16(vl_ + (size_t)d * 2048 + t0 + c, dVl + d * 136 + c); }
    }
  };

  // stage Q planes (async), extract per-wave Q A-fragments (rows wave*16..+15, K=64)
#pragma unroll
  for (int i = 0; i < 4; ++i) {
    int s = i * 256 + tid, r = s >> 3, c = (s & 7) << 3;
    async_copy16(qh_ + (size_t)(qs0 + r) * 64 + c, lQh + r * 72 + c);
    async_copy16(ql_ + (size_t)(qs0 + r) * 64 + c, lQl + r * 72 + c);
  }
  wait_async();
  __syncthreads();
  Frag16 qh[2], ql[2];
  load_afrag16(lQh, wave * 16, 72, 0,  qh[0]);
  load_afrag16(lQh, wave * 16, 72, 32, qh[1]);
  load_afrag16(lQl, wave * 16, 72, 0,  ql[0]);
  load_afrag16(lQl, wave * 16, 72, 32, ql[1]);
  __syncthreads();  // Q frags extracted before P region reuse
  stageKV(0, 0);    // prefetch first K/V tile

  v8f zf = {};
  v8f oacc[4];
#pragma unroll
  for (int i = 0; i < 4; ++i) oacc[i] = zf;
  float mrun[8], lrun[8];
#pragma unroll
  for (int r = 0; r < 8; ++r) { mrun[r] = -INFINITY; lrun[r] = 0.f; }

  int cur = 0;
  for (int t0 = 0; t0 < 2048; t0 += 128) {
    wait_async();
    __syncthreads();  // buf[cur] staged by all waves; buf[cur^1] reads done
    if (t0 + 128 < 2048) stageKV(t0 + 128, cur ^ 1);

    // scores S = (Q/sqrt(D)) @ K^T, per wave: 16 rows x 128 key cols
    const u16* cKh = KH(cur); const u16* cKl = KL(cur);
    v8f sacc[8];
#pragma unroll
    for (int nf = 0; nf < 8; ++nf) sacc[nf] = zf;
#pragma unroll
    for (int kv = 0; kv < 2; ++kv)
#pragma unroll
      for (int nf = 0; nf < 8; ++nf) {
        Frag16 bh_, bl_;
        load_bfrag16(cKh, nf * 16, 72, kv * 32, bh_);
        load_bfrag16(cKl, nf * 16, 72, kv * 32, bl_);
        sacc[nf] = wmma3(qh[kv], ql[kv], bh_, bl_, sacc[nf]);
      }

    // online softmax (rows wave-local; reduce across 16-lane column group)
#pragma unroll
    for (int r = 0; r < 8; ++r) {
      float mx = mrun[r];
#pragma unroll
      for (int nf = 0; nf < 8; ++nf) mx = fmaxf(mx, sacc[nf][r]);
      mx = fmaxf(mx, __shfl_xor(mx, 1, 32));
      mx = fmaxf(mx, __shfl_xor(mx, 2, 32));
      mx = fmaxf(mx, __shfl_xor(mx, 4, 32));
      mx = fmaxf(mx, __shfl_xor(mx, 8, 32));
      float alpha = __expf(mrun[r] - mx);
      mrun[r] = mx;
      lrun[r] *= alpha;
#pragma unroll
      for (int nf2 = 0; nf2 < 4; ++nf2) oacc[nf2][r] *= alpha;
      float rs = 0.f;
#pragma unroll
      for (int nf = 0; nf < 8; ++nf) {
        float p = __expf(sacc[nf][r] - mx);
        sacc[nf][r] = p;
        rs += p;
      }
      rs += __shfl_xor(rs, 1, 32);
      rs += __shfl_xor(rs, 2, 32);
      rs += __shfl_xor(rs, 4, 32);
      rs += __shfl_xor(rs, 8, 32);
      lrun[r] += rs;
    }

    // write P (own 16 rows) packed-interleaved into lP; own-wave LDS ops in-order
#pragma unroll
    for (int nf = 0; nf < 8; ++nf)
#pragma unroll
      for (int r = 0; r < 8; ++r)
        lP[(wave * 16 + r + 8 * (lane >> 4)) * 132 + nf * 16 + (lane & 15)] = pack_hilo(sacc[nf][r]);
    __syncthreads();

    // O += P @ V  (K dim = 128 keys, N = 64 head dims)
    const u16* cVh = VH(cur); const u16* cVl = VL(cur);
#pragma unroll
    for (int kv = 0; kv < 4; ++kv) {
      Frag16 pah, pal;
      load_afrag32(lP, wave * 16, 132, kv * 32, pah, pal);
#pragma unroll
      for (int nf2 = 0; nf2 < 4; ++nf2) {
        Frag16 vbh, vbl;
        load_bfrag16(cVh, nf2 * 16, 136, kv * 32, vbh);
        load_bfrag16(cVl, nf2 * 16, 136, kv * 32, vbl);
        oacc[nf2] = wmma3(pah, pal, vbh, vbl, oacc[nf2]);
      }
    }
    cur ^= 1;
  }

  // epilogue: O/l -> merged-head packed activation y[(b*S+s)*E + h*D + d]
#pragma unroll
  for (int nf2 = 0; nf2 < 4; ++nf2)
#pragma unroll
    for (int r = 0; r < 8; ++r) {
      int srow = qs0 + wave * 16 + r + 8 * (lane >> 4);
      int e = h * 64 + nf2 * 16 + (lane & 15);
      float o = oacc[nf2][r] / lrun[r];
      size_t idx = ((size_t)b * 2048 + srow) * 512 + e;
      u16 h_, l_; split2(o, h_, l_);
      yhi[idx] = h_; ylo[idx] = l_;
    }
}

// ---------- layernorm (faithful: (x-m)/(sqrt(v)+eps)), N=512, 1 wave/row ----------
__global__ __launch_bounds__(256) void layernorm_kernel(
    const float* __restrict__ in, const float* __restrict__ w, const float* __restrict__ bb,
    float* __restrict__ outF, u16* __restrict__ outPh, u16* __restrict__ outPl) {
  int lane = threadIdx.x & 31, wave = threadIdx.x >> 5;
  int row = blockIdx.x * 8 + wave;
  const float* x = in + (size_t)row * 512;
  float v[16];
#pragma unroll
  for (int i = 0; i < 16; ++i) v[i] = x[lane + i * 32];
  float s = 0.f;
#pragma unroll
  for (int i = 0; i < 16; ++i) s += v[i];
#pragma unroll
  for (int m = 1; m < 32; m <<= 1) s += __shfl_xor(s, m, 32);
  float mean = s * (1.0f / 512.0f);
  float s2 = 0.f;
#pragma unroll
  for (int i = 0; i < 16; ++i) { float d = v[i] - mean; s2 += d * d; }
#pragma unroll
  for (int m = 1; m < 32; m <<= 1) s2 += __shfl_xor(s2, m, 32);
  float rstd = 1.0f / (sqrtf(s2 * (1.0f / 512.0f)) + 1e-5f);
#pragma unroll
  for (int i = 0; i < 16; ++i) {
    int c = lane + i * 32;
    float y = (v[i] - mean) * rstd * w[c] + bb[c];
    outF[(size_t)row * 512 + c] = y;
    if (outPh) {
      u16 h_, l_; split2(y, h_, l_);
      outPh[(size_t)row * 512 + c] = h_;
      outPl[(size_t)row * 512 + c] = l_;
    }
  }
}

// ---------- host orchestration ----------
extern "C" void kernel_launch(void* const* d_in, const int* in_sizes, int n_in,
                              void* d_out, int out_size, void* d_ws, size_t ws_size,
                              hipStream_t stream) {
  const float* x    = (const float*)d_in[0];
  const float* WQ   = (const float*)d_in[1];
  const float* bQ   = (const float*)d_in[2];
  const float* WK   = (const float*)d_in[3];
  const float* bK   = (const float*)d_in[4];
  const float* WV   = (const float*)d_in[5];
  const float* bV   = (const float*)d_in[6];
  const float* WY   = (const float*)d_in[7];
  const float* bY   = (const float*)d_in[8];
  const float* ln1w = (const float*)d_in[9];
  const float* ln1b = (const float*)d_in[10];
  const float* W1   = (const float*)d_in[11];
  const float* b1   = (const float*)d_in[12];
  const float* W2   = (const float*)d_in[13];
  const float* b2   = (const float*)d_in[14];
  const float* ln2w = (const float*)d_in[15];
  const float* ln2b = (const float*)d_in[16];

  const int M = 16384, E = 512, F = 2048;
  char* ws = (char*)d_ws;
  size_t off = 0;
  auto alloc = [&](size_t bytes) {
    void* p = ws + off;
    off = (off + bytes + 255) & ~(size_t)255;
    return p;
  };
  auto allocP = [&](size_t n, u16*& hi, u16*& lo) {
    hi = (u16*)alloc(n * 4);
    lo = hi + n;
  };
  u16 *xph, *xpl;   allocP((size_t)M * E, xph, xpl);   // reused as hp after QKV
  u16 *WQth, *WQtl; allocP((size_t)E * E, WQth, WQtl);
  u16 *WKth, *WKtl; allocP((size_t)E * E, WKth, WKtl);
  u16 *WVth, *WVtl; allocP((size_t)E * E, WVth, WVtl);
  u16 *WYth, *WYtl; allocP((size_t)E * E, WYth, WYtl);
  u16 *W1th, *W1tl; allocP((size_t)F * E, W1th, W1tl);
  u16 *W2th, *W2tl; allocP((size_t)E * F, W2th, W2tl);
  u16 *qph, *qpl;   allocP((size_t)M * E, qph, qpl);   // qp..yp region reused as gp
  u16 *kph, *kpl;   allocP((size_t)M * E, kph, kpl);
  u16 *vtph, *vtpl; allocP((size_t)M * E, vtph, vtpl);
  u16 *yph, *ypl;   allocP((size_t)M * E, yph, ypl);
  float* t1  = (float*)alloc((size_t)M * E * 4);       // reused as t2
  float* hbf = (float*)alloc((size_t)M * E * 4);
  u16 *hph = xph, *hpl = xpl;
  u16 *gph = qph, *gpl = qph + (size_t)M * F;          // overlays q/k/vt/y region (134MB)
  float* t2 = t1;
  (void)in_sizes; (void)n_in; (void)out_size; (void)ws_size;

  // pack inputs
  pack_kernel<<<(M * E + 255) / 256, 256, 0, stream>>>(x, xph, xpl, M * E);
  packT_kernel<<<(E * E + 255) / 256, 256, 0, stream>>>(WQ, WQth, WQtl, E, E);
  packT_kernel<<<(E * E + 255) / 256, 256, 0, stream>>>(WK, WKth, WKtl, E, E);
  packT_kernel<<<(E * E + 255) / 256, 256, 0, stream>>>(WV, WVth, WVtl, E, E);
  packT_kernel<<<(E * E + 255) / 256, 256, 0, stream>>>(WY, WYth, WYtl, E, E);
  packT_kernel<<<(E * F + 255) / 256, 256, 0, stream>>>(W1, W1th, W1tl, E, F);
  packT_kernel<<<(F * E + 255) / 256, 256, 0, stream>>>(W2, W2th, W2tl, F, E);

  dim3 gE(M / 128, E / 128), gF(M / 128, F / 128);
  // QKV projections (Q scaled by 1/sqrt(64)); v stored transposed per head
  gemm_bf16x3<EPI_QKV><<<gE, 256, 0, stream>>>(xph, xpl, WQth, WQtl, bQ, nullptr, qph, qpl, M, E, E, 0.125f);
  gemm_bf16x3<EPI_QKV><<<gE, 256, 0, stream>>>(xph, xpl, WKth, WKtl, bK, nullptr, kph, kpl, M, E, E, 1.0f);
  gemm_bf16x3<EPI_VT ><<<gE, 256, 0, stream>>>(xph, xpl, WVth, WVtl, bV, nullptr, vtph, vtpl, M, E, E, 1.0f);
  // flash attention -> merged heads, packed planes
  attn_kernel<<<dim3(16, 64), 256, 0, stream>>>(qph, qpl, kph, kpl, vtph, vtpl, yph, ypl);
  // output projection + residual x
  gemm_bf16x3<EPI_RES><<<gE, 256, 0, stream>>>(yph, ypl, WYth, WYtl, bY, x, t1, nullptr, M, E, E, 1.0f);
  layernorm_kernel<<<M / 8, 256, 0, stream>>>(t1, ln1w, ln1b, hbf, hph, hpl);
  // MLP: gelu(h@W1+b1) @ W2 + b2 + h
  gemm_bf16x3<EPI_GELU><<<gF, 256, 0, stream>>>(hph, hpl, W1th, W1tl, b1, nullptr, gph, gpl, M, F, E, 1.0f);
  gemm_bf16x3<EPI_RES ><<<gE, 256, 0, stream>>>(gph, gpl, W2th, W2tl, b2, hbf, t2, nullptr, M, E, F, 1.0f);
  layernorm_kernel<<<M / 8, 256, 0, stream>>>(t2, ln2w, ln2b, (float*)d_out, nullptr, nullptr);
}
